// HAB_26147760898428
// MI455X (gfx1250) — compile-verified, matrix operand order checked by reading, not run on hardware
//
#include <hip/hip_runtime.h>
#include <hip/hip_bf16.h>
#include <math.h>

typedef _Float16 f16;
typedef __attribute__((ext_vector_type(16))) _Float16 v16h;
typedef __attribute__((ext_vector_type(4)))  _Float16 v4h;
typedef __attribute__((ext_vector_type(8)))  float    v8f;
typedef __attribute__((ext_vector_type(4)))  float    v4f;
typedef __attribute__((ext_vector_type(4)))  unsigned int v4u;

#define HWPIX 65536      // H*W
#define CC    192
#define BB    4
#define NHEAD 6
#define DHEAD 32
#define NTOK  64         // tokens per window
#define NWIN  4096       // B * 32 * 32 windows
#define MROWS (NWIN * NTOK)   // 262144 token rows
#define HIDDEN 768
#define REDC  12
#define RSCALE 0.1f
#define LNEPS 1e-5f

// ---------------------------------------------------------------------------
// WMMA fragment helper (layout per CDNA5 ISA 7.12.2, wave32)
// 16-bit A/W fragment (16x32): lane r=lane&15 holds row r; its 16 halves are
// two contiguous K-runs [hi8, hi8+8) and [16+hi8, 24+hi8), hi8=8*(lane>=16)
//  => exactly two 16-byte loads per lane.
// Requires: ld multiple of 8 halves, base 16B-aligned (true for all callers).
// ---------------------------------------------------------------------------
__device__ __forceinline__ v16h frag_ld_rowK(const f16* p, int ld) {
    union { v16h v; v4u q[2]; } F;
    int lane = threadIdx.x & 31;
    int r    = lane & 15;
    int hi8  = (lane >> 4) << 3;
    const f16* rowp = p + (long)r * ld + hi8;
    F.q[0] = *reinterpret_cast<const v4u*>(rowp);        // K = hi8 .. hi8+7
    F.q[1] = *reinterpret_cast<const v4u*>(rowp + 16);   // K = 16+hi8 .. 23+hi8
    return F.v;
}

__device__ __forceinline__ v8f wmma16(v16h a, v16h b, v8f c) {
    return __builtin_amdgcn_wmma_f32_16x16x32_f16(
        /*neg_a=*/false, a, /*neg_b=*/false, b,
        /*c_mod=*/(short)0, c, /*reuse_a=*/false, /*reuse_b=*/false);
}

// ---------------------------------------------------------------------------
// K0: per-pixel LayerNorm stats over C (mean, rsqrt(var+eps))
// ---------------------------------------------------------------------------
__global__ void ln_stats_kernel(const float* __restrict__ x,
                                float* __restrict__ mOut,
                                float* __restrict__ rsOut) {
    long pid = (long)blockIdx.x * blockDim.x + threadIdx.x;   // B*HW pixels
    if (pid >= (long)BB * HWPIX) return;
    int b = (int)(pid / HWPIX);
    int p = (int)(pid % HWPIX);
    const float* xp = x + (long)b * CC * HWPIX + p;
    float s = 0.f, s2 = 0.f;
    for (int c = 0; c < CC; ++c) {
        float v = xp[(long)c * HWPIX];
        s += v; s2 += v * v;
    }
    float m   = s * (1.f / CC);
    float var = s2 * (1.f / CC) - m * m;
    mOut[pid]  = m;
    rsOut[pid] = rsqrtf(var + LNEPS);
}

// ---------------------------------------------------------------------------
// K1: global average pool of LN1(x) per (b,c)
// ---------------------------------------------------------------------------
__global__ void pool_kernel(const float* __restrict__ x,
                            const float* __restrict__ m,
                            const float* __restrict__ rs,
                            const float* __restrict__ n1w,
                            const float* __restrict__ n1b,
                            float* __restrict__ y) {
    __shared__ float red[256];
    int bc = blockIdx.x;              // 0 .. B*C-1
    int b  = bc / CC, c = bc % CC;
    const float* xp = x  + (long)bc * HWPIX;
    const float* mp = m  + (long)b * HWPIX;
    const float* rp = rs + (long)b * HWPIX;
    float s = 0.f;
    for (int p = threadIdx.x; p < HWPIX; p += blockDim.x)
        s += (xp[p] - mp[p]) * rp[p];
    red[threadIdx.x] = s;
    __syncthreads();
    for (int st = 128; st > 0; st >>= 1) {
        if ((int)threadIdx.x < st) red[threadIdx.x] += red[threadIdx.x + st];
        __syncthreads();
    }
    if (threadIdx.x == 0)
        y[bc] = n1w[c] * (red[0] * (1.f / HWPIX)) + n1b[c];
}

// ---------------------------------------------------------------------------
// K2: channel-attention SE MLP (tiny, one block)
// ---------------------------------------------------------------------------
__global__ void ca_mlp_kernel(const float* __restrict__ y,
                              const float* __restrict__ w1,
                              const float* __restrict__ w2,
                              float* __restrict__ sOut) {
    __shared__ float yl[BB * CC];
    __shared__ float tl[BB * REDC];
    for (int i = threadIdx.x; i < BB * CC; i += blockDim.x) yl[i] = y[i];
    __syncthreads();
    if ((int)threadIdx.x < BB * REDC) {
        int b = threadIdx.x / REDC, r = threadIdx.x % REDC;
        float s = 0.f;
        for (int c = 0; c < CC; ++c) s += w1[r * CC + c] * yl[b * CC + c];
        tl[threadIdx.x] = fmaxf(s, 0.f);
    }
    __syncthreads();
    for (int i = threadIdx.x; i < BB * CC; i += blockDim.x) {
        int b = i / CC, c = i % CC;
        float s = 0.f;
        for (int r = 0; r < REDC; ++r) s += w2[c * REDC + r] * tl[b * REDC + r];
        sOut[i] = 1.f / (1.f + __expf(-s));
    }
}

// ---------------------------------------------------------------------------
// K3: x1 = LN1(x)*sigmoid_scale*RS + x
// ---------------------------------------------------------------------------
__global__ void ca_apply_kernel(const float* __restrict__ x,
                                const float* __restrict__ m,
                                const float* __restrict__ rs,
                                const float* __restrict__ n1w,
                                const float* __restrict__ n1b,
                                const float* __restrict__ sG,
                                float* __restrict__ x1) {
    long i = (long)blockIdx.x * blockDim.x + threadIdx.x;
    if (i >= (long)BB * CC * HWPIX) return;
    int  p  = (int)(i % HWPIX);
    long bc = i / HWPIX;
    int  c  = (int)(bc % CC);
    int  b  = (int)(bc / CC);
    long bp = (long)b * HWPIX + p;
    float xv = x[i];
    float xc = (xv - m[bp]) * rs[bp] * n1w[c] + n1b[c];
    x1[i] = xc * sG[bc] * RSCALE + xv;
}

// ---------------------------------------------------------------------------
// K4: LN2 + window partition -> f16 token matrix [MROWS, C]
// ---------------------------------------------------------------------------
__global__ void win_ln2_kernel(const float* __restrict__ x1,
                               const float* __restrict__ m2,
                               const float* __restrict__ rs2,
                               const float* __restrict__ n2w,
                               const float* __restrict__ n2b,
                               f16* __restrict__ xw) {
    long i = (long)blockIdx.x * blockDim.x + threadIdx.x;   // MROWS*C
    if (i >= (long)MROWS * CC) return;
    int  c   = (int)(i % CC);
    long row = i / CC;
    int  n   = (int)(row % NTOK);
    int  win = (int)(row / NTOK);
    int  b   = win >> 10;
    int  rem = win & 1023;
    int  wr  = rem >> 5, wc = rem & 31;
    int  h   = wr * 8 + (n >> 3);
    int  w   = wc * 8 + (n & 7);
    long p   = (long)h * 256 + w;
    long bp  = (long)b * HWPIX + p;
    float v  = x1[((long)(b * CC + c)) * HWPIX + p];
    xw[i] = (f16)((v - m2[bp]) * rs2[bp] * n2w[c] + n2b[c]);
}

// ---------------------------------------------------------------------------
// K5: f32 -> f16 conversion (weights)
// ---------------------------------------------------------------------------
__global__ void cvt_f16_kernel(const float* __restrict__ src,
                               f16* __restrict__ dst, long n) {
    long i = (long)blockIdx.x * blockDim.x + threadIdx.x;
    if (i < n) dst[i] = (f16)src[i];
}

// ---------------------------------------------------------------------------
// K6: TRANSPOSED relative position bias table  biasT[head][kj][qi]
// (transposed so the attention kernel can load 8 qi-contiguous values
//  with two b128 loads per tile instead of 8 strided b32 gathers)
// ---------------------------------------------------------------------------
__global__ void rpb_kernel(const float* __restrict__ rpb,
                           float* __restrict__ biasT) {
    int i = blockIdx.x * blockDim.x + threadIdx.x;
    if (i >= NHEAD * NTOK * NTOK) return;
    int head = i / (NTOK * NTOK);
    int rem  = i % (NTOK * NTOK);
    int kj   = rem >> 6, qi = rem & 63;
    int d0   = (qi >> 3) - (kj >> 3) + 7;
    int d1   = (qi & 7)  - (kj & 7)  + 7;
    biasT[i] = rpb[(d0 * 15 + d1) * NHEAD + head];
}

// ---------------------------------------------------------------------------
// K7: register-blocked WMMA GEMM  out[M,N] = A[M,K] @ W[N,K]^T (+ epilogue)
// Each wave computes a 64x32 output block (4 M-tiles x 2 N-tiles).
// EPI 0: QKV -> q,k stored f16 row-major; V slice stored TRANSPOSED into
//        outH2[win][head][d][key] (f16) so P@V can use frag_ld_rowK.
// EPI 1: +bias -> f32 (proj)
// EPI 2: +bias, GELU -> f16 (ffn1)
// EPI 3: extra + (acc+bias)*rs -> f32 (ffn2)
// ---------------------------------------------------------------------------
template <int EPI>
__global__ __launch_bounds__(256) void gemm_wmma_kernel(
    const f16* __restrict__ A, const f16* __restrict__ W,
    const float* __restrict__ bias, const float* __restrict__ extra,
    float* __restrict__ outF, f16* __restrict__ outH, f16* __restrict__ outH2,
    int M, int N, int K, float rs) {
    int  wave   = threadIdx.x >> 5;
    long wtile  = (long)blockIdx.x * (blockDim.x >> 5) + wave;
    int  nWN    = N >> 5;                       // 32-wide wave tiles in N
    long nWaves = (long)(M >> 6) * nWN;         // 64-tall wave tiles in M
    if (wtile >= nWaves) return;
    int wm = (int)(wtile / nWN);
    int wn = (int)(wtile % nWN);

    v8f acc[4][2];
#pragma unroll
    for (int i = 0; i < 4; ++i)
#pragma unroll
        for (int j = 0; j < 2; ++j)
            acc[i][j] = (v8f){0.f, 0.f, 0.f, 0.f, 0.f, 0.f, 0.f, 0.f};

    const f16* Ap = A + (long)wm * 64 * K;
    const f16* Wp = W + (long)wn * 32 * K;
    for (int k0 = 0; k0 < K; k0 += 32) {
        if (k0 + 32 < K)   // prefetch next A k-slab (global_prefetch_b8)
            __builtin_prefetch(Ap + ((threadIdx.x & 31) & 15) * K + k0 + 32, 0, 3);
        v16h a[4], b[2];
#pragma unroll
        for (int tm = 0; tm < 4; ++tm)
            a[tm] = frag_ld_rowK(Ap + (long)tm * 16 * K + k0, K);
#pragma unroll
        for (int tn = 0; tn < 2; ++tn)
            b[tn] = frag_ld_rowK(Wp + (long)tn * 16 * K + k0, K);
#pragma unroll
        for (int tm = 0; tm < 4; ++tm)
#pragma unroll
            for (int tn = 0; tn < 2; ++tn)
                acc[tm][tn] = wmma16(a[tm], b[tn], acc[tm][tn]);
    }

    int lane = threadIdx.x & 31;
    int n15  = lane & 15;
    int hi8  = (lane >> 4) << 3;
#pragma unroll
    for (int tn = 0; tn < 2; ++tn) {
        int col  = wn * 32 + tn * 16 + n15;
        float bv = (EPI == 0) ? 0.f : bias[col];
#pragma unroll
        for (int tm = 0; tm < 4; ++tm) {
#pragma unroll
            for (int r = 0; r < 8; ++r) {
                long row = (long)wm * 64 + tm * 16 + r + hi8;
                float v  = acc[tm][tn][r];
                if (EPI == 0) {
                    if (col < 384) {   // q,k: row-major
                        __builtin_nontemporal_store((f16)v, outH + row * 576 + col);
                    } else {           // v: transposed per (win,head)
                        int  hd  = (col - 384) >> 5;
                        int  dd  = (col - 384) & 31;
                        long win = row >> 6, key = row & 63;
                        long vix = (((win * NHEAD + hd) << 5) + dd) * 64 + key;
                        __builtin_nontemporal_store((f16)v, outH2 + vix);
                    }
                } else if (EPI == 1) {
                    __builtin_nontemporal_store(v + bv, outF + row * N + col);
                } else if (EPI == 2) {
                    float xg = v + bv;
                    f16 g = (f16)(0.5f * xg * (1.f + erff(xg * 0.70710678118654752f)));
                    __builtin_nontemporal_store(g, outH + row * N + col);
                } else {
                    long idx = row * N + col;
                    __builtin_nontemporal_store(extra[idx] + (v + bv) * rs, outF + idx);
                }
            }
        }
    }
}

// ---------------------------------------------------------------------------
// K8: window attention — one wave per (window, head)
// Q·K^T (16 WMMAs, K=32) -> softmax(+biasT) in LDS -> P·V^T (16 WMMAs, K=64)
// ---------------------------------------------------------------------------
__global__ __launch_bounds__(64) void attn_kernel(
    const f16* __restrict__ qkv,     // [MROWS, 576] (q,k valid)
    const f16* __restrict__ vT,      // [NWIN,NHEAD,32,64]
    const float* __restrict__ biasT, // [NHEAD][kj][qi]
    f16* __restrict__ o16) {         // [MROWS, 192]
    __shared__ __align__(16) float Slog[2][NTOK * 68];
    __shared__ __align__(16) f16   Sp[2][NTOK * 72];

    int  wid  = threadIdx.x >> 5;
    int  lane = threadIdx.x & 31;
    long task = (long)blockIdx.x * 2 + wid;   // NWIN*NHEAD tasks
    if (task >= (long)NWIN * NHEAD) return;
    int win  = (int)(task / NHEAD);
    int head = (int)(task % NHEAD);

    const f16* qb  = qkv + (long)win * NTOK * 576 + head * DHEAD;
    const f16* kb  = qb + 192;
    const f16* vbT = vT + ((long)win * NHEAD + head) * DHEAD * NTOK;
    float* logits  = Slog[wid];
    f16*   P       = Sp[wid];
    const float scale = 0.17677669529663687f;   // 1/sqrt(32)

    int n15 = lane & 15;
    int hi8 = (lane >> 4) << 3;

    // ---- Q fragments (4 M-tiles), hoisted ----
    v16h aq[4];
#pragma unroll
    for (int tm = 0; tm < 4; ++tm)
        aq[tm] = frag_ld_rowK(qb + (long)tm * 16 * 576, 576);

    // ---- logits = scale * Q K^T + bias ----
#pragma unroll
    for (int tn = 0; tn < 4; ++tn) {
        v16h bk = frag_ld_rowK(kb + (long)tn * 16 * 576, 576);
        int  kj = tn * 16 + n15;
#pragma unroll
        for (int tm = 0; tm < 4; ++tm) {
            v8f c = {0.f, 0.f, 0.f, 0.f, 0.f, 0.f, 0.f, 0.f};
            c = wmma16(aq[tm], bk, c);
            // transposed bias: 8 qi-contiguous floats = 2 x b128
            const float* bp = biasT + head * 4096 + kj * 64 + tm * 16 + hi8;
            v4f b0 = *reinterpret_cast<const v4f*>(bp);
            v4f b1 = *reinterpret_cast<const v4f*>(bp + 4);
#pragma unroll
            for (int r = 0; r < 8; ++r) {
                int qi = tm * 16 + r + hi8;
                float bb = (r < 4) ? b0[r] : b1[r - 4];
                logits[qi * 68 + kj] = c[r] * scale + bb;
            }
        }
    }

    // ---- softmax (each lane owns 2 rows, vectorized float4/half4) ----
#pragma unroll
    for (int rr = 0; rr < 2; ++rr) {
        int qi = lane + rr * 32;
        v4f* rowp = reinterpret_cast<v4f*>(logits + qi * 68);
        float mx = -1e30f;
        for (int j = 0; j < 16; ++j) {
            v4f t = rowp[j];
            mx = fmaxf(mx, fmaxf(fmaxf(t[0], t[1]), fmaxf(t[2], t[3])));
        }
        float sum = 0.f;
        for (int j = 0; j < 16; ++j) {
            v4f t = rowp[j];
            v4f e = {__expf(t[0] - mx), __expf(t[1] - mx),
                     __expf(t[2] - mx), __expf(t[3] - mx)};
            rowp[j] = e;
            sum += e[0] + e[1] + e[2] + e[3];
        }
        float inv = 1.f / sum;
        v4h* pp = reinterpret_cast<v4h*>(P + qi * 72);
        for (int j = 0; j < 16; ++j) {
            v4f t = rowp[j];
            v4h h = {(f16)(t[0] * inv), (f16)(t[1] * inv),
                     (f16)(t[2] * inv), (f16)(t[3] * inv)};
            pp[j] = h;
        }
    }

    // ---- O = P @ V  (V stored transposed: vT[d][key], A@W^T pattern) ----
    v16h bv[2][2];
#pragma unroll
    for (int tn = 0; tn < 2; ++tn)
#pragma unroll
        for (int ks = 0; ks < 2; ++ks)
            bv[tn][ks] = frag_ld_rowK(vbT + (long)tn * 16 * 64 + ks * 32, 64);

#pragma unroll
    for (int tm = 0; tm < 4; ++tm) {
#pragma unroll
        for (int tn = 0; tn < 2; ++tn) {
            v8f c = {0.f, 0.f, 0.f, 0.f, 0.f, 0.f, 0.f, 0.f};
#pragma unroll
            for (int ks = 0; ks < 2; ++ks) {
                v16h a = frag_ld_rowK(P + tm * 16 * 72 + ks * 32, 72);
                c = wmma16(a, bv[tn][ks], c);
            }
#pragma unroll
            for (int r = 0; r < 8; ++r) {
                long row = (long)win * NTOK + tm * 16 + r + hi8;
                int  col = head * DHEAD + tn * 16 + n15;
                o16[row * CC + col] = (f16)c[r];
            }
        }
    }
}

// ---------------------------------------------------------------------------
// K9: LN3 per token row (wave per row, wave32 shfl reduce)
// ---------------------------------------------------------------------------
__global__ void ln3_kernel(const float* __restrict__ pin,
                           const float* __restrict__ w,
                           const float* __restrict__ bparm,
                           float* __restrict__ out32,
                           f16* __restrict__ out16) {
    int  wid  = threadIdx.x >> 5;
    int  lane = threadIdx.x & 31;
    long row  = (long)blockIdx.x * (blockDim.x >> 5) + wid;
    if (row >= (long)MROWS) return;
    const float* rp = pin + row * CC;
    float v[6];
    float s = 0.f, s2 = 0.f;
#pragma unroll
    for (int j = 0; j < 6; ++j) {
        v[j] = rp[lane + j * 32];
        s += v[j]; s2 += v[j] * v[j];
    }
#pragma unroll
    for (int o = 16; o > 0; o >>= 1) {
        s  += __shfl_xor(s, o, 32);
        s2 += __shfl_xor(s2, o, 32);
    }
    float m  = s * (1.f / CC);
    float rs = rsqrtf(s2 * (1.f / CC) - m * m + LNEPS);
#pragma unroll
    for (int j = 0; j < 6; ++j) {
        int c = lane + j * 32;
        float o = (v[j] - m) * rs * w[c] + bparm[c];
        out32[row * CC + c] = o;
        out16[row * CC + c] = (f16)o;
    }
}

// ---------------------------------------------------------------------------
// K10: un-window + final residual: out = final_tok*RS + x1
// ---------------------------------------------------------------------------
__global__ void final_kernel(const float* __restrict__ fin,
                             const float* __restrict__ x1,
                             float* __restrict__ out) {
    long i = (long)blockIdx.x * blockDim.x + threadIdx.x;
    if (i >= (long)BB * CC * HWPIX) return;
    int  p  = (int)(i % HWPIX);
    long bc = i / HWPIX;
    int  c  = (int)(bc % CC);
    int  b  = (int)(bc / CC);
    int  h  = p >> 8, w = p & 255;
    long row = ((long)(b * 1024 + (h >> 3) * 32 + (w >> 3))) * NTOK +
               (h & 7) * 8 + (w & 7);
    out[i] = fin[row * CC + c] * RSCALE + x1[i];
}

// ---------------------------------------------------------------------------
// launch
// ---------------------------------------------------------------------------
extern "C" void kernel_launch(void* const* d_in, const int* in_sizes, int n_in,
                              void* d_out, int out_size, void* d_ws, size_t ws_size,
                              hipStream_t stream) {
    const float* x      = (const float*)d_in[0];
    const float* n1w    = (const float*)d_in[1];
    const float* n1b    = (const float*)d_in[2];
    const float* n2w    = (const float*)d_in[3];
    const float* n2b    = (const float*)d_in[4];
    const float* n3w    = (const float*)d_in[5];
    const float* n3b    = (const float*)d_in[6];
    const float* ca_w1  = (const float*)d_in[7];
    const float* ca_w2  = (const float*)d_in[8];
    const float* qkv_w  = (const float*)d_in[9];
    const float* proj_w = (const float*)d_in[10];
    const float* proj_b = (const float*)d_in[11];
    const float* rpb    = (const float*)d_in[12];
    const float* ffn_w1 = (const float*)d_in[13];
    const float* ffn_b1 = (const float*)d_in[14];
    const float* ffn_w2 = (const float*)d_in[15];
    const float* ffn_b2 = (const float*)d_in[16];
    float* out = (float*)d_out;

    char*  ws  = (char*)d_ws;
    size_t off = 0;
    auto alloc = [&](size_t bytes) -> char* {
        char* p = ws + off;
        off += (bytes + 255) & ~(size_t)255;
        return p;
    };
    const size_t NEL = (size_t)BB * CC * HWPIX;          // 50,331,648
    float* x1      = (float*)alloc(NEL * 4);
    f16*   xw16    = (f16*)  alloc(NEL * 2);             // reused as o16
    f16*   qkv16   = (f16*)  alloc((size_t)MROWS * 576 * 2);
    float* proj32  = (float*)alloc((size_t)MROWS * CC * 4);
    float* oln32   = (float*)alloc((size_t)MROWS * CC * 4);
    f16*   oln16   = (f16*)  alloc((size_t)MROWS * CC * 2);
    float* fin32   = (float*)alloc((size_t)MROWS * CC * 4);
    f16*   vT16    = (f16*)  alloc((size_t)NWIN * NHEAD * DHEAD * NTOK * 2);
    float* mbuf    = (float*)alloc((size_t)BB * HWPIX * 4);
    float* rsbuf   = (float*)alloc((size_t)BB * HWPIX * 4);
    float* ybuf    = (float*)alloc(BB * CC * 4);
    float* sbuf    = (float*)alloc(BB * CC * 4);
    float* biasT   = (float*)alloc(NHEAD * NTOK * NTOK * 4);
    f16*   qkvw16  = (f16*)alloc((size_t)576 * CC * 2);
    f16*   projw16 = (f16*)alloc((size_t)CC * CC * 2);
    f16*   fw1_16  = (f16*)alloc((size_t)HIDDEN * CC * 2);
    f16*   fw2_16  = (f16*)alloc((size_t)CC * HIDDEN * 2);
    f16*   h16     = (f16*)qkv16;   // alias: qkv16+proj32 region, both dead by ffn1
    f16*   o16     = xw16;          // alias: xw16 dead after QKV GEMM

    const int T = 256;
    long nPix = (long)BB * HWPIX;

    // --- channel attention branch ---
    ln_stats_kernel<<<(nPix + T - 1) / T, T, 0, stream>>>(x, mbuf, rsbuf);
    pool_kernel<<<BB * CC, T, 0, stream>>>(x, mbuf, rsbuf, n1w, n1b, ybuf);
    ca_mlp_kernel<<<1, T, 0, stream>>>(ybuf, ca_w1, ca_w2, sbuf);
    ca_apply_kernel<<<(long)(NEL + T - 1) / T, T, 0, stream>>>(
        x, mbuf, rsbuf, n1w, n1b, sbuf, x1);

    // --- LN2 + window partition ---
    ln_stats_kernel<<<(nPix + T - 1) / T, T, 0, stream>>>(x1, mbuf, rsbuf);
    win_ln2_kernel<<<(long)(NEL + T - 1) / T, T, 0, stream>>>(
        x1, mbuf, rsbuf, n2w, n2b, xw16);

    // --- weight conversion + bias table ---
    cvt_f16_kernel<<<(576 * CC + T - 1) / T, T, 0, stream>>>(qkv_w, qkvw16, 576 * CC);
    cvt_f16_kernel<<<(CC * CC + T - 1) / T, T, 0, stream>>>(proj_w, projw16, CC * CC);
    cvt_f16_kernel<<<(HIDDEN * CC + T - 1) / T, T, 0, stream>>>(ffn_w1, fw1_16, HIDDEN * CC);
    cvt_f16_kernel<<<(CC * HIDDEN + T - 1) / T, T, 0, stream>>>(ffn_w2, fw2_16, CC * HIDDEN);
    rpb_kernel<<<(NHEAD * NTOK * NTOK + T - 1) / T, T, 0, stream>>>(rpb, biasT);

    // --- QKV GEMM: [M,192] @ [576,192]^T -> q,k f16 + transposed V ---
    {
        long waves = (long)(MROWS / 64) * (576 / 32);
        gemm_wmma_kernel<0><<<(waves + 7) / 8, T, 0, stream>>>(
            xw16, qkvw16, nullptr, nullptr, nullptr, qkv16, vT16,
            MROWS, 576, CC, 0.f);
    }

    // --- windowed attention ---
    {
        long tasks = (long)NWIN * NHEAD;
        attn_kernel<<<(tasks + 1) / 2, 64, 0, stream>>>(qkv16, vT16, biasT, o16);
    }

    // --- projection: [M,192] @ [192,192]^T + b -> f32 ---
    {
        long waves = (long)(MROWS / 64) * (CC / 32);
        gemm_wmma_kernel<1><<<(waves + 7) / 8, T, 0, stream>>>(
            o16, projw16, proj_b, nullptr, proj32, nullptr, nullptr,
            MROWS, CC, CC, 0.f);
    }

    // --- LN3 ---
    ln3_kernel<<<(MROWS + 7) / 8, T, 0, stream>>>(proj32, n3w, n3b, oln32, oln16);

    // --- FFN1: [M,192] @ [768,192]^T + b1, GELU -> f16 ---
    {
        long waves = (long)(MROWS / 64) * (HIDDEN / 32);
        gemm_wmma_kernel<2><<<(waves + 7) / 8, T, 0, stream>>>(
            oln16, fw1_16, ffn_b1, nullptr, nullptr, h16, nullptr,
            MROWS, HIDDEN, CC, 0.f);
    }

    // --- FFN2: oln + ([M,768] @ [192,768]^T + b2)*RS -> f32 ---
    {
        long waves = (long)(MROWS / 64) * (CC / 32);
        gemm_wmma_kernel<3><<<(waves + 7) / 8, T, 0, stream>>>(
            h16, fw2_16, ffn_b2, oln32, fin32, nullptr, nullptr,
            MROWS, CC, HIDDEN, RSCALE);
    }

    // --- un-window + final residual ---
    final_kernel<<<(long)(NEL + T - 1) / T, T, 0, stream>>>(fin32, x1, out);
}
